// TemporalMamba_2731599200652
// MI455X (gfx1250) — matrix operand order repaired and checked by reading, not compiled, not run on hardware
//
#include <hip/hip_runtime.h>
#include <hip/hip_bf16.h>
#include <stdint.h>

#define DEV __device__ __forceinline__

typedef __bf16 bf16_t;
typedef __attribute__((ext_vector_type(16))) __bf16 v16bf;
typedef __attribute__((ext_vector_type(8)))  float  v8f;
typedef __attribute__((ext_vector_type(4)))  unsigned int v4u;
typedef __attribute__((ext_vector_type(8)))  int    v8i;
typedef __attribute__((ext_vector_type(4)))  int    v4i;

// Problem constants (from reference)
constexpr int BB     = 2;
constexpr int LL     = 1024;
constexpr int DMODEL = 1024;
constexpr int DINNER = 2048;          // EXPAND * D_MODEL
constexpr int DSTATE = 16;
constexpr int DTRANK = 64;            // D_MODEL / 16
constexpr int XPROJ_N = DTRANK + 2 * DSTATE;   // 96

// TDM GEMM tiling
constexpr int KC         = 64;                  // K-chunk staged per TDM load
constexpr int TILE_ROWS  = 64;                  // N rows per staged B tile
constexpr int ROW_PITCH  = KC * 2 + 16;         // 144B: 128B data + 16B TDM pad
constexpr int TILE_BYTES = TILE_ROWS * ROW_PITCH;   // 9216B per buffer

DEV bf16_t f2bf(float f) {
  unsigned u = __builtin_bit_cast(unsigned, f);
  unsigned r = u + 0x7FFFu + ((u >> 16) & 1u);   // round-to-nearest-even
  unsigned short h = (unsigned short)(r >> 16);
  return __builtin_bit_cast(bf16_t, h);
}

DEV float softplusf(float x) {
  return (x > 20.f) ? x : __logf(1.f + __expf(x));
}

// ---------------------------------------------------------------------------
// Elementwise cast helpers
// ---------------------------------------------------------------------------
__global__ void cast_bf16_kernel(const float* __restrict__ in,
                                 bf16_t* __restrict__ outp, int n) {
  int i = blockIdx.x * blockDim.x + threadIdx.x;
  if (i < n) outp[i] = f2bf(in[i]);
}

// in: R x C (f32, row-major) -> out: C x R (bf16, row-major)  == B^T for GEMM
__global__ void cast_transpose_bf16_kernel(const float* __restrict__ in,
                                           bf16_t* __restrict__ outp,
                                           int R, int Ccols) {
  int i = blockIdx.x * blockDim.x + threadIdx.x;
  if (i >= R * Ccols) return;
  int r = i / Ccols, c = i % Ccols;
  outp[(size_t)c * R + r] = f2bf(in[i]);
}

// dt_lr slice: x_dbl[:, 0:64] (stride 96) -> contiguous bf16 (B*L x 64)
__global__ void slice_cast_kernel(const float* __restrict__ xdbl,
                                  bf16_t* __restrict__ outp) {
  int i = blockIdx.x * blockDim.x + threadIdx.x;   // over B*L*DTRANK
  int r = i >> 6, c = i & 63;
  outp[i] = f2bf(xdbl[(size_t)r * XPROJ_N + c]);
}

// ---------------------------------------------------------------------------
// Tensor Data Mover: stage a TILE_ROWS x KC bf16 tile of Bt (N x K row-major)
// into LDS at `lds_addr`, with 16B padding every row (ROW_PITCH = 144B) so
// ds_load_b128 fragment reads are 16B-aligned and bank-conflict-free.
// Descriptor per CDNA5 ISA (groups 0/1; 2D tensor, later
// groups zero). 6-arg clang-23 builtin form; tracked by TENSORcnt.
// ---------------------------------------------------------------------------
DEV void tdm_load_tile(const bf16_t* gsrc, unsigned lds_addr, int K, int Nrows) {
  uint64_t ga = (uint64_t)(uintptr_t)gsrc;
  v4u g0;
  g0[0] = 1u;                                            // count=1, user mode
  g0[1] = lds_addr;                                      // LDS byte address
  g0[2] = (unsigned)ga;                                  // global_addr[31:0]
  g0[3] = (unsigned)((ga >> 32) & 0x01FFFFFFu)           // global_addr[56:32]
        | (2u << 30);                                    // type = 2 ("image")
  v8i g1;
  g1[0] = (int)((1u << 16)      // data_size = 2 bytes (bf16)
              | (1u << 20)      // pad_enable
              | (4u << 22)      // pad_interval: 32 dwords = 128B (one row)
              | (3u << 25));    // pad_amount: 4 dwords = 16B
  g1[1] = (int)(((unsigned)K & 0xFFFFu) << 16);          // tensor_dim0[15:0]
  g1[2] = (int)((((unsigned)K >> 16) & 0xFFFFu)          // tensor_dim0[31:16]
              | (((unsigned)Nrows & 0xFFFFu) << 16));    // tensor_dim1[15:0]
  g1[3] = (int)(((unsigned)KC & 0xFFFFu) << 16);         // tile_dim0 = KC
  g1[4] = (int)TILE_ROWS;                                // tile_dim1 = 64 rows
  g1[5] = (int)K;                                        // tensor_dim0_stride lo
  g1[6] = 0;                                             // stride hi / dim1_stride lo
  g1[7] = 0;
  v4i z4 = {};
  v8i z8 = {};
  __builtin_amdgcn_tensor_load_to_lds(g0, g1, z4, z4, z8, 0);
}

struct Frag { union { v16bf v; v4u q[2]; }; };

// ---------------------------------------------------------------------------
// TDM-staged WMMA bf16 GEMM:  C(f32, MxN) = A(bf16 MxK) x Bt^T (Bt: NxK bf16)
// Block = 256 threads (8 waves): block tile 128(M) x 64(N).
// Each wave: one 16-row strip x 64 cols = 4 accumulators, A fragment reused 4x.
// B tile (64 rows x KC) double-buffered in LDS via tensor_load_to_lds; the DMA
// of chunk k+1 overlaps WMMA on chunk k. mode 1: softplus(acc + bias[n]).
// Requires: M % 128 == 0, N % 64 == 0, K % 64 == 0.
// ---------------------------------------------------------------------------
__global__ void wmma_gemm_tdm_kernel(const bf16_t* __restrict__ A,
                                     const bf16_t* __restrict__ Bt,
                                     float* __restrict__ C,
                                     const float* __restrict__ bias,
                                     int M, int N, int K, int mode) {
  __shared__ __align__(16) char smem[2 * TILE_BYTES];
  const int lane = threadIdx.x & 31;
  const int wave = threadIdx.x >> 5;
  const int m0   = blockIdx.y * 128 + wave * 16;
  const int n0   = blockIdx.x * 64;

  // A fragment (16-bit A 16x32): lanes 0-15 hold K {0..7,16..23},
  // lanes 16-31 hold K {8..15,24..31} of row M = m0 + lane%16.
  const int asel = (lane & 16) ? 8 : 0;
  // B fragment (16-bit B 32x16): lanes 0-15 hold K {0..7,16..23}, lanes 16-31
  // K {8..15,24..31} of column N = lane%16 (same order as A).
  const int bsel_b = (lane & 16) ? 16 : 0;               // bytes within row

  const bf16_t* arow = A + (size_t)(m0 + (lane & 15)) * K;
  const unsigned lds_base = (unsigned)(uintptr_t)smem;

  v8f acc[4];
#pragma unroll
  for (int f = 0; f < 4; ++f) acc[f] = (v8f){};

  // Prologue: wave 0 kicks off the first B-tile DMA.
  if (threadIdx.x < 32)
    tdm_load_tile(Bt + (size_t)n0 * K, lds_base, K, N);

  int buf = 0;
  for (int kc = 0; kc < K; kc += KC) {
    if (threadIdx.x < 32)
      __builtin_amdgcn_s_wait_tensorcnt(0);   // current tile landed in LDS
    __syncthreads();
    const int cur = buf;
    if (kc + KC < K) {                         // overlap next DMA with compute
      if (threadIdx.x < 32)
        tdm_load_tile(Bt + (size_t)n0 * K + kc + KC,
                      lds_base + (unsigned)((buf ^ 1) * TILE_BYTES), K, N);
      buf ^= 1;
    }
    const char* tb = smem + cur * TILE_BYTES;

#pragma unroll
    for (int s = 0; s < 2; ++s) {              // two 16x16x32 steps per chunk
      const int kk = kc + s * 32;
      Frag a;
      a.q[0] = *(const v4u*)(arow + kk + asel);
      a.q[1] = *(const v4u*)(arow + kk + 16 + asel);
#pragma unroll
      for (int f = 0; f < 4; ++f) {
        const int roff = (f * 16 + (lane & 15)) * ROW_PITCH + s * 64 + bsel_b;
        Frag b;
        b.q[0] = *(const v4u*)(tb + roff);
        b.q[1] = *(const v4u*)(tb + roff + 32);
        acc[f] = __builtin_amdgcn_wmma_f32_16x16x32_bf16(
            false, a.v, false, b.v, (short)0, acc[f], false, false);
      }
    }
    __syncthreads();                           // all reads done before reuse
  }

  // C/D layout: VGPR r -> row m0 + r (+8 for lanes 16-31), col = n-tile + lane%16
  const int rbase = m0 + ((lane >> 4) << 3);
#pragma unroll
  for (int f = 0; f < 4; ++f) {
    const int col = n0 + f * 16 + (lane & 15);
#pragma unroll
    for (int r = 0; r < 8; ++r) {
      float v = acc[f][r];
      if (mode == 1) v = softplusf(v + bias[col]);
      C[(size_t)(rbase + r) * N + col] = v;
    }
  }
}

// ---------------------------------------------------------------------------
// Direct-load WMMA GEMM (for N=96, not a multiple of the TDM tile width).
// Block = 128 threads (4 waves); each wave: 16 rows x 32 cols.
// ---------------------------------------------------------------------------
__global__ void wmma_gemm_kernel(const bf16_t* __restrict__ A,
                                 const bf16_t* __restrict__ Bt,
                                 float* __restrict__ C,
                                 const float* __restrict__ bias,
                                 int M, int N, int K, int mode) {
  const int lane = threadIdx.x & 31;
  const int wave = threadIdx.x >> 5;
  const int m0 = (blockIdx.y * 4 + wave) * 16;
  const int n0 = blockIdx.x * 32;
  if (m0 >= M) return;

  const int asel = (lane & 16) ? 8 : 0;
  const int bsel = (lane & 16) ? 8 : 0;

  const bf16_t* arow  = A  + (size_t)(m0 + (lane & 15)) * K;
  const bf16_t* b0row = Bt + (size_t)(n0 + (lane & 15)) * K;
  const bf16_t* b1row = Bt + (size_t)(n0 + 16 + (lane & 15)) * K;

  v8f c0 = {}; v8f c1 = {};
  for (int k = 0; k < K; k += 32) {
    if (k + 128 < K) {
      __builtin_prefetch(arow  + k + 128, 0, 1);
      __builtin_prefetch(b0row + k + 128, 0, 1);
      __builtin_prefetch(b1row + k + 128, 0, 1);
    }
    Frag a, b0, b1;
    a.q[0]  = *(const v4u*)(arow  + k + asel);
    a.q[1]  = *(const v4u*)(arow  + k + 16 + asel);
    b0.q[0] = *(const v4u*)(b0row + k + bsel);
    b0.q[1] = *(const v4u*)(b0row + k + bsel + 16);
    b1.q[0] = *(const v4u*)(b1row + k + bsel);
    b1.q[1] = *(const v4u*)(b1row + k + bsel + 16);
    c0 = __builtin_amdgcn_wmma_f32_16x16x32_bf16(false, a.v, false, b0.v,
                                                 (short)0, c0, false, false);
    c1 = __builtin_amdgcn_wmma_f32_16x16x32_bf16(false, a.v, false, b1.v,
                                                 (short)0, c1, false, false);
  }

  const int col0  = n0 + (lane & 15);
  const int rbase = m0 + ((lane >> 4) << 3);
#pragma unroll
  for (int r = 0; r < 8; ++r) {
    float v0 = c0[r], v1 = c1[r];
    if (mode == 1) {
      v0 = softplusf(v0 + bias[col0]);
      v1 = softplusf(v1 + bias[col0 + 16]);
    }
    C[(size_t)(rbase + r) * N + col0]      = v0;
    C[(size_t)(rbase + r) * N + col0 + 16] = v1;
  }
}

// ---------------------------------------------------------------------------
// Causal depthwise conv (k=4) + bias + SiLU.  Reads the x_in half of xz.
// ---------------------------------------------------------------------------
__global__ void conv_silu_kernel(const float* __restrict__ xz,
                                 const float* __restrict__ conv_w,
                                 const float* __restrict__ conv_b,
                                 float* __restrict__ u_f32,
                                 bf16_t* __restrict__ u_bf) {
  int i = blockIdx.x * blockDim.x + threadIdx.x;   // over B*L*DINNER
  int d  = i & (DINNER - 1);
  int bl = i >> 11;                                // DINNER == 2^11
  int l  = bl & (LL - 1);
  const size_t stride = 2 * DINNER;
  size_t base = (size_t)bl * stride + d;           // x_in = columns [0,2048)
  float acc = conv_b[d];
  float w0 = conv_w[d * 4 + 0], w1 = conv_w[d * 4 + 1];
  float w2 = conv_w[d * 4 + 2], w3 = conv_w[d * 4 + 3];
  if (l >= 3) acc += xz[base - 3 * stride] * w0;
  if (l >= 2) acc += xz[base - 2 * stride] * w1;
  if (l >= 1) acc += xz[base - 1 * stride] * w2;
  acc += xz[base] * w3;
  float uu = acc / (1.f + __expf(-acc));           // silu
  u_f32[i] = uu;
  u_bf[i]  = f2bf(uu);
}

// ---------------------------------------------------------------------------
// Selective-scan: one thread per (b, d) channel; h[16] in registers.
// Fuses dA/dBu materialization, D_skip add and silu(z) gate.
// ---------------------------------------------------------------------------
__global__ void scan_kernel(const float* __restrict__ xz,
                            const float* __restrict__ x_dbl,
                            const float* __restrict__ delta,
                            const float* __restrict__ u,
                            const float* __restrict__ A_log,
                            const float* __restrict__ D_skip,
                            bf16_t* __restrict__ y_bf) {
  __shared__ float sBC[32];
  const int blocksPerB = DINNER / 256;             // 8
  const int b = blockIdx.x / blocksPerB;
  const int d = (blockIdx.x % blocksPerB) * 256 + threadIdx.x;

  float Ad[DSTATE];
#pragma unroll
  for (int n = 0; n < DSTATE; ++n)
    Ad[n] = -__expf(A_log[(size_t)d * DSTATE + n]);
  float h[DSTATE];
#pragma unroll
  for (int n = 0; n < DSTATE; ++n) h[n] = 0.f;
  const float dsk = D_skip[d];

  for (int l = 0; l < LL; ++l) {
    const size_t row = (size_t)b * LL + l;
    __syncthreads();
    if (threadIdx.x < 32)
      sBC[threadIdx.x] = x_dbl[row * XPROJ_N + DTRANK + threadIdx.x];
    __syncthreads();

    const float dt = delta[row * DINNER + d];
    const float uu = u[row * DINNER + d];
    const float zz = xz[row * (2 * DINNER) + DINNER + d];   // z half
    const float du = dt * uu;
    float y = 0.f;
#pragma unroll
    for (int n = 0; n < DSTATE; ++n) {
      h[n] = __expf(dt * Ad[n]) * h[n] + du * sBC[n];       // B
      y   += h[n] * sBC[16 + n];                            // C
    }
    y += dsk * uu;
    y *= zz / (1.f + __expf(-zz));                          // * silu(z)
    y_bf[row * DINNER + d] = f2bf(y);
  }
}

// ---------------------------------------------------------------------------
extern "C" void kernel_launch(void* const* d_in, const int* in_sizes, int n_in,
                              void* d_out, int out_size, void* d_ws, size_t ws_size,
                              hipStream_t stream) {
  const float* x       = (const float*)d_in[0];
  const float* W_in    = (const float*)d_in[1];
  const float* conv_w  = (const float*)d_in[2];
  const float* conv_b  = (const float*)d_in[3];
  const float* W_xproj = (const float*)d_in[4];
  const float* W_dt    = (const float*)d_in[5];
  const float* b_dt    = (const float*)d_in[6];
  const float* A_log   = (const float*)d_in[7];
  const float* D_skip  = (const float*)d_in[8];
  const float* W_out   = (const float*)d_in[9];
  float* out = (float*)d_out;
  (void)in_sizes; (void)n_in; (void)out_size; (void)ws_size;

  char* p = (char*)d_ws;
  size_t off = 0;
  auto carve = [&](size_t bytes) -> void* {
    void* r = p + off;
    off += (bytes + 255) & ~(size_t)255;
    return r;
  };
  const size_t BL = (size_t)BB * LL;
  bf16_t* x_bf    = (bf16_t*)carve(BL * DMODEL * 2);
  bf16_t* win_t   = (bf16_t*)carve((size_t)DMODEL * 2 * DINNER * 2);
  float*  xz      = (float*) carve(BL * 2 * DINNER * 4);
  float*  u_f32   = (float*) carve(BL * DINNER * 4);
  bf16_t* u_bf    = (bf16_t*)carve(BL * DINNER * 2);
  bf16_t* wxp_t   = (bf16_t*)carve((size_t)XPROJ_N * DINNER * 2);
  float*  x_dbl   = (float*) carve(BL * XPROJ_N * 4);
  bf16_t* dtlr    = (bf16_t*)carve(BL * DTRANK * 2);
  bf16_t* wdt_t   = (bf16_t*)carve((size_t)DINNER * DTRANK * 2);
  float*  delta_f = (float*) carve(BL * DINNER * 4);
  bf16_t* y_bf    = (bf16_t*)carve(BL * DINNER * 2);
  bf16_t* wout_t  = (bf16_t*)carve((size_t)DMODEL * DINNER * 2);

  // 1) Cast operands for GEMM 1
  cast_bf16_kernel<<<(BL * DMODEL) / 256, 256, 0, stream>>>(x, x_bf, BL * DMODEL);
  cast_transpose_bf16_kernel<<<((size_t)DMODEL * 2 * DINNER) / 256, 256, 0, stream>>>(
      W_in, win_t, DMODEL, 2 * DINNER);
  // 2) xz = x @ W_in   (M=2048, N=4096, K=1024) — TDM-staged WMMA
  {
    dim3 g(2 * DINNER / 64, BL / 128);
    wmma_gemm_tdm_kernel<<<g, 256, 0, stream>>>(x_bf, win_t, xz, nullptr,
                                                (int)BL, 2 * DINNER, DMODEL, 0);
  }
  // 3) depthwise conv + SiLU -> u
  conv_silu_kernel<<<(BL * DINNER) / 256, 256, 0, stream>>>(xz, conv_w, conv_b,
                                                            u_f32, u_bf);
  // 4) x_dbl = u @ W_xproj   (M=2048, N=96, K=2048) — direct-load WMMA
  cast_transpose_bf16_kernel<<<((size_t)DINNER * XPROJ_N) / 256, 256, 0, stream>>>(
      W_xproj, wxp_t, DINNER, XPROJ_N);
  {
    dim3 g(XPROJ_N / 32, BL / 64);
    wmma_gemm_kernel<<<g, 128, 0, stream>>>(u_bf, wxp_t, x_dbl, nullptr,
                                            (int)BL, XPROJ_N, DINNER, 0);
  }
  // 5) delta = softplus(dt_lr @ W_dt + b_dt)   (M=2048, N=2048, K=64)
  slice_cast_kernel<<<(BL * DTRANK) / 256, 256, 0, stream>>>(x_dbl, dtlr);
  cast_transpose_bf16_kernel<<<((size_t)DTRANK * DINNER) / 256, 256, 0, stream>>>(
      W_dt, wdt_t, DTRANK, DINNER);
  {
    dim3 g(DINNER / 64, BL / 128);
    wmma_gemm_tdm_kernel<<<g, 256, 0, stream>>>(dtlr, wdt_t, delta_f, b_dt,
                                                (int)BL, DINNER, DTRANK, 1);
  }
  // 6) fused selective scan + skip + gate -> y (bf16)
  scan_kernel<<<BB * (DINNER / 256), 256, 0, stream>>>(xz, x_dbl, delta_f,
                                                       u_f32, A_log, D_skip, y_bf);
  // 7) out = y @ W_out   (M=2048, N=1024, K=2048)
  cast_transpose_bf16_kernel<<<((size_t)DINNER * DMODEL) / 256, 256, 0, stream>>>(
      W_out, wout_t, DINNER, DMODEL);
  {
    dim3 g(DMODEL / 64, BL / 128);
    wmma_gemm_tdm_kernel<<<g, 256, 0, stream>>>(y_bf, wout_t, out, nullptr,
                                                (int)BL, DMODEL, DINNER, 0);
  }
}
